// Controller_90142773609330
// MI455X (gfx1250) — matrix operand order repaired and checked by reading, not compiled, not run on hardware
//
#include <hip/hip_runtime.h>

// ---------------- problem constants (from reference) ----------------
#define NUM_OPS  14
#define NUM_MAGS 10
#define Q_SUB    5
#define L_OPS    4
#define EMB      32
#define HID      100
#define BATCH    8192
#define NSTEP    (Q_SUB * L_OPS * 2)   // 40

// ---------------- tiling ----------------
#define KPAD          160              // 32 (x) + 128 (h padded)  -> 5 K-chunks of 32
#define UPAD          112              // 100 hidden units padded to 7 tiles of 16
#define WROWS         448              // 4 gates * 112
#define ROWS_PER_WAVE 16
#define WAVES_PER_WG  4
#define ROWS_PER_WG   (ROWS_PER_WAVE * WAVES_PER_WG)

typedef __attribute__((ext_vector_type(16))) _Float16 v16h;
typedef __attribute__((ext_vector_type(8)))  _Float16 v8h;
typedef __attribute__((ext_vector_type(8)))  float    v8f;
typedef __attribute__((ext_vector_type(4)))  unsigned u32x4;
typedef __attribute__((ext_vector_type(8)))  int      i32x8;
typedef __attribute__((ext_vector_type(4)))  int      i32x4;

// ---------------- static LDS image: identical layout in ws and LDS ----------------
#define SM_W      0                              // f16 [448][160]            143360
#define SM_BIAS   (WROWS * KPAD * 2)             // f32 [448]        @143360    1792
#define SM_EMB    (SM_BIAS + WROWS * 4)          // f16 [24][32]     @145152    1536
#define SM_WOP    (SM_EMB + 24 * EMB * 2)        // f32 [14][100]    @146688    5600
#define SM_BOP    (SM_WOP  + NUM_OPS  * HID * 4) // f32 [16]         @152288      64
#define SM_WMAG   (SM_BOP  + 16 * 4)             // f32 [10][100]    @152352    4000
#define SM_BMAG   (SM_WMAG + NUM_MAGS * HID * 4) // f32 [16]         @156352      64
#define SM_IMG    (SM_BMAG + 16 * 4)             // image size = 156416 B (39104 dw)
// dynamic LDS beyond the image
#define SM_ABUF   SM_IMG                         // f16 [4 waves][16][160] = 20480
#define SM_PACC   (SM_ABUF + WAVES_PER_WG * ROWS_PER_WAVE * KPAD * 2)
#define SM_BYTES  (SM_PACC + 24 * 4)
// workspace = image + global prob accumulators
#define WS_GACC   SM_IMG
#define WS_END    (WS_GACC + 24 * 4)

#define TDM_CHUNK_B  (SM_IMG / 4)                // 39104 bytes per descriptor
#define TDM_CHUNK_DW (SM_IMG / 16)               // 9776 dwords per descriptor

// ---------------- helpers ----------------
__device__ __forceinline__ float fsig(float x)  { return 1.0f / (1.0f + __expf(-x)); }
__device__ __forceinline__ float ftanh(float x) { float e = __expf(2.0f * x); return (e - 1.0f) / (e + 1.0f); }
__device__ __forceinline__ unsigned mix32(unsigned x) {
  x ^= x >> 16; x *= 0x7feb352du;
  x ^= x >> 15; x *= 0x846ca68bu;
  x ^= x >> 16; return x;
}

// ---------------- Tensor Data Mover issue (literal cpol at each site) ----------------
// NOTE: __has_builtin() for amdgcn builtins is only true during the DEVICE pass
// of HIP's dual compilation; inside __global__ bodies the builtins themselves are
// accepted in both passes (aux-target builtins). So only #error on the device pass.
#if defined(__HIP_DEVICE_COMPILE__) && !__has_builtin(__builtin_amdgcn_tensor_load_to_lds)
#error "gfx1250 __builtin_amdgcn_tensor_load_to_lds not available on this toolchain"
#endif

#if defined(__clang_major__) && (__clang_major__ >= 23)
#define TDM_CALL(G0, G1, Z4) do {                                         \
    i32x8 _z8 = {0, 0, 0, 0, 0, 0, 0, 0};                                 \
    __builtin_amdgcn_tensor_load_to_lds((G0), (G1), (Z4), (Z4), _z8, 0);  \
  } while (0)
#else
#define TDM_CALL(G0, G1, Z4)                                              \
    __builtin_amdgcn_tensor_load_to_lds((G0), (G1), (Z4), (Z4), 0)
#endif

// 1-D dword-granularity tile load: NDW dwords global(GA) -> LDS(LDSOFF)
#define TDM_LOAD_1D(GA, LDSOFF, NDW) do {                                 \
    u32x4 _g0;                                                            \
    _g0[0] = 1u;                                  /* count=1, user */     \
    _g0[1] = (unsigned)(LDSOFF);                  /* lds_addr */          \
    _g0[2] = (unsigned)(GA);                      /* global_addr[31:0] */ \
    _g0[3] = ((unsigned)((GA) >> 32) & 0x01FFFFFFu) | (2u << 30);         \
    i32x8 _g1;                                                            \
    _g1[0] = (int)(2u << 16);                     /* data_size = 4B */    \
    _g1[1] = (int)(((NDW) & 0xFFFFu) << 16);      /* tensor_dim0 lo */    \
    _g1[2] = (int)(((NDW) >> 16) & 0xFFFFu);      /* tensor_dim0 hi */    \
    _g1[3] = (int)(((NDW) & 0xFFFFu) << 16);      /* tile_dim0 */         \
    _g1[4] = 0;                                   /* tile_dim1/2 unused */\
    _g1[5] = (int)(NDW);                          /* dim0 stride lo */    \
    _g1[6] = 0; _g1[7] = 0;                                               \
    i32x4 _z4 = {0, 0, 0, 0};                                             \
    TDM_CALL(_g0, _g1, _z4);                                              \
  } while (0)

// ======================================================================
// prep: build the full LDS image in ws (f16 WMMA weights in
// [gate-major, unit-padded, K-padded] layout, fused bias, f16 embedding,
// classifier weights) + zeroed global prob accumulators.
// ======================================================================
__global__ void prep_kernel(const float* __restrict__ emb,
                            const float* __restrict__ Wih,
                            const float* __restrict__ Whh,
                            const float* __restrict__ bih,
                            const float* __restrict__ bhh,
                            const float* __restrict__ Wop,
                            const float* __restrict__ bop,
                            const float* __restrict__ Wmag,
                            const float* __restrict__ bmag,
                            char* __restrict__ ws) {
  int idx = blockIdx.x * blockDim.x + threadIdx.x;
  _Float16* Wst  = (_Float16*)(ws + SM_W);
  float*    bst  = (float*)(ws + SM_BIAS);
  _Float16* est  = (_Float16*)(ws + SM_EMB);
  float*    gacc = (float*)(ws + WS_GACC);

  if (idx < WROWS * KPAD) {
    int sr = idx / KPAD, col = idx % KPAD;
    int g = sr / UPAD,   u   = sr % UPAD;
    float v = 0.0f;
    if (u < HID) {
      int src = g * HID + u;
      if (col < EMB)            v = Wih[src * EMB + col];
      else if (col < EMB + HID) v = Whh[src * HID + (col - EMB)];
    }
    Wst[idx] = (_Float16)v;
  }
  if (idx < WROWS) {
    int g = idx / UPAD, u = idx % UPAD;
    float v = 0.0f;
    if (u < HID) { int src = g * HID + u; v = bih[src] + bhh[src]; }
    bst[idx] = v;
  }
  if (idx < 24 * EMB) est[idx] = (_Float16)emb[idx];
  if (idx < NUM_OPS  * HID) ((float*)(ws + SM_WOP))[idx]  = Wop[idx];
  if (idx < NUM_MAGS * HID) ((float*)(ws + SM_WMAG))[idx] = Wmag[idx];
  if (idx < 16) ((float*)(ws + SM_BOP))[idx]  = (idx < NUM_OPS)  ? bop[idx]  : 0.0f;
  if (idx < 16) ((float*)(ws + SM_BMAG))[idx] = (idx < NUM_MAGS) ? bmag[idx] : 0.0f;
  if (idx < 24) gacc[idx] = 0.0f;
}

// ======================================================================
// main controller kernel: 1 wave = 16 batch rows, all 40 steps local.
// ======================================================================
__global__ __launch_bounds__(WAVES_PER_WG * 32)
void ctrl_kernel(const char* __restrict__ ws,
                 float* __restrict__ gaccG, float* __restrict__ out) {
  extern __shared__ char sm[];
  const int tid = threadIdx.x;

  // ---- stage the whole static image global -> LDS via TDM ----
  // Four descriptors = four outstanding TENSORcnt DMA streams per WG.
  if (tid < 32) {
    unsigned long long ga = (unsigned long long)ws;
    unsigned lo =
        (unsigned)(unsigned long long)(__attribute__((address_space(3))) const char*)sm;
    TDM_LOAD_1D(ga + 0ull * TDM_CHUNK_B, lo + 0u * TDM_CHUNK_B, TDM_CHUNK_DW);
    TDM_LOAD_1D(ga + 1ull * TDM_CHUNK_B, lo + 1u * TDM_CHUNK_B, TDM_CHUNK_DW);
    TDM_LOAD_1D(ga + 2ull * TDM_CHUNK_B, lo + 2u * TDM_CHUNK_B, TDM_CHUNK_DW);
    TDM_LOAD_1D(ga + 3ull * TDM_CHUNK_B, lo + 3u * TDM_CHUNK_B, TDM_CHUNK_DW);
    __builtin_amdgcn_s_wait_tensorcnt(0);
  }
  {
    float* pacc0 = (float*)(sm + SM_PACC);
    if (tid < 24) pacc0[tid] = 0.0f;
  }
  __syncthreads();

  const _Float16* Wl    = (const _Float16*)(sm + SM_W);
  const float*    biasl = (const float*)(sm + SM_BIAS);
  const _Float16* embl  = (const _Float16*)(sm + SM_EMB);
  float*          pacc  = (float*)(sm + SM_PACC);

  const int wave = tid >> 5;
  const int lane = tid & 31;
  const int nIdx = lane & 15;   // N within tile == local batch row for scalar stages
  const int hl   = lane >> 4;   // wave half (K-split / M-split per ISA layouts)
  _Float16* ab = (_Float16*)(sm + SM_ABUF) + wave * ROWS_PER_WAVE * KPAD;
  const int rowBase = blockIdx.x * ROWS_PER_WG + wave * ROWS_PER_WAVE;
  const int rowG = rowBase + nIdx;

  float lpSum = 0.0f, entSum = 0.0f;
  float cst[7][8];

#pragma unroll 1
  for (int q = 0; q < Q_SUB; ++q) {
    // reset h, c, x for this sub-policy
#pragma unroll
    for (int u7 = 0; u7 < 7; ++u7)
#pragma unroll
      for (int j = 0; j < 8; ++j) cst[u7][j] = 0.0f;
    {
      v8h z = {};
      v8h* abv = (v8h*)ab;
      for (int i = lane; i < ROWS_PER_WAVE * KPAD / 8; i += 32) abv[i] = z;
    }

#pragma unroll 1
    for (int j4 = 0; j4 < L_OPS; ++j4) {
#pragma unroll 1
      for (int kind = 0; kind < 2; ++kind) {
        const int s = (q * L_OPS + j4) * 2 + kind;

        // ---------- LSTM gates: 16x400 = A[16x160] * W^T via WMMA ----------
        // A operand (16-bit 16x32 layout): lanes 0-15 row M=lane K{0..7,16..23},
        // lanes 16-31 row M=lane-16 K{8..15,24..31}.
        v16h a[5];
#pragma unroll
        for (int kc = 0; kc < 5; ++kc) {
          const v8h* lo = (const v8h*)(ab + nIdx * KPAD + kc * 32 + hl * 8);
          const v8h* hi = (const v8h*)(ab + nIdx * KPAD + kc * 32 + hl * 8 + 16);
          v8h l = *lo, h = *hi;
          a[kc] = __builtin_shufflevector(l, h, 0, 1, 2, 3, 4, 5, 6, 7,
                                               8, 9, 10, 11, 12, 13, 14, 15);
        }

#pragma unroll
        for (int ut = 0; ut < 7; ++ut) {
          v8f acc[4];
#pragma unroll
          for (int g = 0; g < 4; ++g) {
            float bv = biasl[g * UPAD + ut * 16 + nIdx];
            const int wrow = g * UPAD + ut * 16 + nIdx;
            v8f acv;
#pragma unroll
            for (int t = 0; t < 8; ++t) acv[t] = bv;
            // B operand (32x16): lane n<16 -> N=n K{0..7,16..23}; lane>=16 -> N, K{8..15,24..31}.
#pragma unroll
            for (int kc = 0; kc < 5; ++kc) {
              const v8h* b0 = (const v8h*)(Wl + wrow * KPAD + kc * 32 + hl * 8);
              const v8h* b1 = (const v8h*)(Wl + wrow * KPAD + kc * 32 + hl * 8 + 16);
              v8h bl = *b0, bh = *b1;
              v16h bo = __builtin_shufflevector(bl, bh, 0, 1, 2, 3, 4, 5, 6, 7,
                                                        8, 9, 10, 11, 12, 13, 14, 15);
              acv = __builtin_amdgcn_wmma_f32_16x16x32_f16(
                  false, a[kc], false, bo, (short)0, acv, false, false);
            }
            acc[g] = acv;
          }
          // cell update; gates lane-aligned in C layout (VGPR j = row j / j+8)
#pragma unroll
          for (int j = 0; j < 8; ++j) {
            float iv = fsig(acc[0][j]), fv = fsig(acc[1][j]);
            float gv = ftanh(acc[2][j]), ov = fsig(acc[3][j]);
            float c2 = fv * cst[ut][j] + iv * gv;
            cst[ut][j] = c2;
            float h2 = ov * ftanh(c2);
            int rr = j + hl * 8;
            ab[rr * KPAD + EMB + ut * 16 + nIdx] = (_Float16)h2;
          }
        }

        // ---------- logits / softmax / entropy / Gumbel sample ----------
        const float* Wc = kind ? (const float*)(sm + SM_WMAG) : (const float*)(sm + SM_WOP);
        const float* bc = kind ? (const float*)(sm + SM_BMAG) : (const float*)(sm + SM_BOP);
        const int K = kind ? NUM_MAGS : NUM_OPS;

        float zloc[7];
        float m = -1e30f;
        const _Float16* hrow = ab + nIdx * KPAD + EMB;
#pragma unroll
        for (int ci = 0; ci < 7; ++ci) {
          int c = hl + ci * 2;
          float z = -1e30f;
          if (c < K) {
            float dot = bc[c];
#pragma unroll 4
            for (int u = 0; u < HID; ++u) dot += (float)hrow[u] * Wc[c * HID + u];
            z = 2.5f * ftanh(dot);
          }
          zloc[ci] = z;
          m = fmaxf(m, z);
        }
        m = fmaxf(m, __shfl_xor(m, 16));

        float se = 0.0f, sz = 0.0f;
        float bestV = -1e30f, bestZ = 0.0f;
        int bestI = 0;
#pragma unroll
        for (int ci = 0; ci < 7; ++ci) {
          int c = hl + ci * 2;
          if (c < K) {
            float e = __expf(zloc[ci] - m);
            se += e;
            sz += zloc[ci] * e;
            unsigned hsh = mix32((unsigned)(rowG * NSTEP + s) * 1664525u +
                                 (unsigned)c * 1013904223u + 0x9e3779b9u);
            float u01 = ((float)hsh + 0.5f) * (1.0f / 4294967296.0f);
            float gum = -__logf(-__logf(u01));
            float val = zloc[ci] + gum;
            if (val > bestV) { bestV = val; bestZ = zloc[ci]; bestI = c; }
          }
        }
        se += __shfl_xor(se, 16);
        sz += __shfl_xor(sz, 16);
        {
          float oV = __shfl_xor(bestV, 16);
          float oZ = __shfl_xor(bestZ, 16);
          int   oI = __shfl_xor(bestI, 16);
          if (oV > bestV) { bestV = oV; bestZ = oZ; bestI = oI; }
        }
        float lse = m + __logf(se);
        entSum += lse - sz / se;          // entropy
        lpSum  += bestZ - lse;            // selected log-prob
        if (hl == 0) out[(long)rowG * NSTEP + s] = (float)bestI;

        // probability accumulation (mean over steps x batch, finalized later)
#pragma unroll
        for (int ci = 0; ci < 7; ++ci) {
          int c = hl + ci * 2;
          if (c < K) atomicAdd(&pacc[kind * NUM_OPS + c], __expf(zloc[ci] - lse));
        }

        // ---------- next input: embedding lookup ----------
        const int eoff = kind ? NUM_OPS : 0;
        {
          const v8h* er = (const v8h*)(embl + (eoff + bestI) * EMB + hl * 16);
          v8h e0 = er[0], e1 = er[1];
          v8h* aw = (v8h*)(ab + nIdx * KPAD + hl * 16);
          aw[0] = e0; aw[1] = e1;
        }
      }
    }
  }

  if (hl == 0) {
    out[(long)BATCH * NSTEP + 24 + rowG]         = lpSum;
    out[(long)BATCH * NSTEP + 24 + BATCH + rowG] = entSum;
  }
  __syncthreads();
  if (tid < 24) atomicAdd(&gaccG[tid], pacc[tid]);
}

// ======================================================================
// finalize: scale prob sums to means
// ======================================================================
__global__ void fin_kernel(const float* __restrict__ gacc, float* __restrict__ out) {
  int t = threadIdx.x;
  const float inv = 1.0f / ((float)(Q_SUB * L_OPS) * (float)BATCH);
  if (t < 24) out[(long)BATCH * NSTEP + t] = gacc[t] * inv;
}

// ======================================================================
extern "C" void kernel_launch(void* const* d_in, const int* in_sizes, int n_in,
                              void* d_out, int out_size, void* d_ws, size_t ws_size,
                              hipStream_t stream) {
  const float* emb  = (const float*)d_in[0];
  const float* Wih  = (const float*)d_in[1];
  const float* Whh  = (const float*)d_in[2];
  const float* bih  = (const float*)d_in[3];
  const float* bhh  = (const float*)d_in[4];
  const float* Wop  = (const float*)d_in[5];
  const float* bop  = (const float*)d_in[6];
  const float* Wmag = (const float*)d_in[7];
  const float* bmag = (const float*)d_in[8];
  char*  ws   = (char*)d_ws;
  float* out  = (float*)d_out;
  float* gacc = (float*)(ws + WS_GACC);

  const int prepN = WROWS * KPAD;
  prep_kernel<<<(prepN + 255) / 256, 256, 0, stream>>>(
      emb, Wih, Whh, bih, bhh, Wop, bop, Wmag, bmag, ws);
  ctrl_kernel<<<BATCH / ROWS_PER_WG, WAVES_PER_WG * 32, SM_BYTES, stream>>>(ws, gacc, out);
  fin_kernel<<<1, 32, 0, stream>>>(gacc, out);
}
